// GatedLinearRecurrence_54640573939866
// MI455X (gfx1250) — matrix operand order repaired and checked by reading, not compiled, not run on hardware
//
#include <hip/hip_runtime.h>
#include <math.h>

// ---------------- problem constants ----------------
#define B_    2
#define S_    2048
#define DM_   1024
#define H_    16
#define DH_   64
#define HID_  1024
#define MTOK_ (B_ * S_)     // 4096 tokens
#define NC_   (S_ / 64)     // 32 chunks of length 64
#define LDH   72            // f16 LDS row stride (144B -> 16B aligned, matches TDM pad)

typedef __attribute__((ext_vector_type(16))) _Float16 v16h;
typedef __attribute__((ext_vector_type(8)))  _Float16 v8h;
typedef __attribute__((ext_vector_type(8)))  float    v8f;
typedef __attribute__((ext_vector_type(4)))  unsigned int v4u;
typedef __attribute__((ext_vector_type(8)))  int      v8i;
typedef __attribute__((ext_vector_type(4)))  int      v4i;

// ---------------- WMMA tile loaders (wave32 layouts, CDNA5 ISA 7.12.2) ----
__device__ __forceinline__ v16h mk16(v8h lo, v8h hi) {
  v16h r;
#pragma unroll
  for (int i = 0; i < 8; ++i) { r[i] = lo[i]; r[8 + i] = hi[i]; }
  return r;
}

// A tile 16x32 f16: lane L: row m = L&15, half hh = L>>4.
__device__ __forceinline__ v16h load_a16x32(const _Float16* base, int lda, int lane) {
  const int m = lane & 15, hh = lane >> 4;
  const _Float16* p = base + (size_t)m * lda + hh * 8;
  v8h lo = *(const v8h*)p;
  v8h hi = *(const v8h*)(p + 16);
  return mk16(lo, hi);
}

// B tile 32x16 f16 where B[k][n] = BT[n][k], BT row-major with stride ldb.
__device__ __forceinline__ v16h load_b16x32(const _Float16* baseT, int ldb, int lane) {
  const int n = lane & 15, hh = lane >> 4;
  const _Float16* p = baseT + (size_t)n * ldb + hh * 8;
  v8h lo = *(const v8h*)p;
  v8h hi = *(const v8h*)(p + 16);
  return mk16(lo, hi);
}

#define WMMA_F16(a, b, c) \
  __builtin_amdgcn_wmma_f32_16x16x32_f16(false, (a), false, (b), (short)0, (c), false, false)

// ---------------- TDM descriptors (CDNA5 ISA, async_tensor ch. 7-8) ----
// 2D tile: 64 x 64 f16 elements, global row stride 1024 elements, LDS padded
// 4 DWORDs every 32 DWORDs -> 144B LDS row stride == LDH halfs.
__device__ __forceinline__ v4u tdm_group0(const void* gptr, unsigned lds_off) {
  union { v4u v; unsigned long long q[2]; unsigned int w[4]; } u;
  u.w[0] = 1u;       // count=1 (valid descriptor), user mode, no gather
  u.w[1] = lds_off;  // lds_addr (bytes, relative to wave LDS base)
  unsigned long long a = (unsigned long long)(uintptr_t)gptr;
  u.q[1] = (a & ((1ULL << 57) - 1)) | (2ULL << 62);  // global_addr | type=2 (image)
  return u.v;
}

__device__ __forceinline__ v8i tdm_group1() {
  v8i g;
  g[0] = (int)((1u << 16)      // data_size = 1 -> 2 bytes
             | (1u << 20)      // pad_enable
             | (4u << 22)      // pad_interval: 4 -> every 32 DWORDs (128B)
             | (3u << 25));    // pad_amount:   3 -> 4 DWORDs (16B)
  g[1] = (int)(64u << 16);     // tensor_dim0 = 64   (bits 79:48, low half)
  g[2] = (int)(64u << 16);     // tensor_dim1 = 64   (bits 111:80, low half)
  g[3] = (int)(64u << 16);     // tile_dim0   = 64   (bits 127:112)
  g[4] = 64;                   // tile_dim1   = 64   (bits 143:128)
  g[5] = 1024;                 // tensor_dim0_stride = 1024 elems (bits 207:160)
  g[6] = 0;
  g[7] = 0;
  return g;
}

__device__ __forceinline__ void tdm_load_tile(const void* gptr, unsigned lds_off) {
  v4u g0 = tdm_group0(gptr, lds_off);
  v8i g1 = tdm_group1();
  v4i z4 = {0, 0, 0, 0};
  v8i z8 = {0, 0, 0, 0, 0, 0, 0, 0};
  // 6-arg form (probe-confirmed by this toolchain's diagnostic)
  __builtin_amdgcn_tensor_load_to_lds(g0, g1, z4, z4, z8, 0);
}

// ---------------- f32 -> f16 conversion ----------------
__global__ void cvt_f32_to_f16_kernel(const float* __restrict__ in,
                                      _Float16* __restrict__ out, int n) {
  int i = blockIdx.x * blockDim.x + threadIdx.x;
  if (i < n) out[i] = (_Float16)in[i];
}

// ---------------- WMMA GEMM: C[M,N] = A[M,K] * W[N,K]^T ----------------
template <bool OF16>
__global__ __launch_bounds__(256) void wmma_gemm_kernel(
    const _Float16* __restrict__ A, const _Float16* __restrict__ W,
    float* __restrict__ Cf, _Float16* __restrict__ Ch, int M, int N, int K) {
  const int wave = threadIdx.x >> 5, lane = threadIdx.x & 31;
  const int n0 = blockIdx.x * 64;
  const int m0 = blockIdx.y * 128 + wave * 16;
  v8f acc[4] = {};
  for (int k0 = 0; k0 < K; k0 += 32) {
    if (k0 + 64 < K) {  // global_prefetch_b8 of next A block
      __builtin_prefetch(A + (size_t)m0 * K + k0 + 64, 0, 1);
    }
    v16h a = load_a16x32(A + (size_t)m0 * K + k0, K, lane);
#pragma unroll
    for (int j = 0; j < 4; ++j) {
      if (n0 + 16 * j < N) {
        v16h b = load_b16x32(W + (size_t)(n0 + 16 * j) * K + k0, K, lane);
        acc[j] = WMMA_F16(a, b, acc[j]);
      }
    }
  }
  const int n = lane & 15, hh = lane >> 4;
#pragma unroll
  for (int j = 0; j < 4; ++j) {
    const int col = n0 + 16 * j + n;
    if (col < N) {
#pragma unroll
      for (int r = 0; r < 8; ++r) {
        const size_t o = (size_t)(m0 + hh * 8 + r) * N + col;
        if constexpr (OF16) Ch[o] = (_Float16)acc[j][r];
        else                Cf[o] = acc[j][r];
      }
    }
  }
}

// ---------------- per-chunk decay prep ----------------
__global__ __launch_bounds__(64) void gla_prep_kernel(
    _Float16* __restrict__ qh, _Float16* __restrict__ kh,
    const _Float16* __restrict__ gh, const float* __restrict__ dproj,
    const float* __restrict__ bd, float* __restrict__ Asum) {
  const int idx = blockIdx.x;
  const int c = idx & (NC_ - 1);
  const int h = (idx >> 5) & (H_ - 1);
  const int b = idx >> 9;
  const int t = threadIdx.x;
  __shared__ float gsh[64];

  const int s = c * 64 + t;
  const float draw = dproj[((size_t)b * S_ + s) * H_ + h] + bd[h];
  const float sp = (draw > 20.f) ? draw : log1pf(__expf(draw));
  gsh[t] = -sp;
  __syncthreads();

  float Acum = 0.f;
  for (int i = 0; i <= t; ++i) Acum += gsh[i];
  if (t == 63) Asum[idx] = Acum;  // chunk-total decay A_L

  const float eA = __expf(Acum), enA = __expf(-Acum);
  const size_t base = (((size_t)b * S_ + s) * H_ + h) * DH_;
#pragma unroll 4
  for (int d = 0; d < DH_; ++d) {
    float qv = (float)qh[base + d];
    qh[base + d] = (_Float16)(qv * eA);
    float kv = (float)kh[base + d];
    float gv = (float)gh[base + d];
    float sig = 1.f / (1.f + __expf(-gv));
    kh[base + d] = (_Float16)(kv * sig * enA);
  }
}

// ---------------- chunked GLA scan (1 WG per (b,h), 8 waves, TDM staging) ----
// Dynamic LDS layout (element offsets, LDH = 72 halfs per row):
#define EQ2  (0 * 64 * LDH)  // [t][dk]   TDM
#define EK2  (1 * 64 * LDH)  // [t][dk]   TDM
#define EK2T (2 * 64 * LDH)  // [dk][t]
#define EVT  (3 * 64 * LDH)  // [dv][t]
#define ESC  (4 * 64 * LDH)  // [t][j] masked scores (also TDM temp for V rows)
#define ES0T (5 * 64 * LDH)  // state transposed [dv][dk]
#define SCAN_LDS_BYTES (6 * 64 * LDH * 2)

__global__ __launch_bounds__(256) void gla_scan_kernel(
    const _Float16* __restrict__ qh, const _Float16* __restrict__ kh,
    const _Float16* __restrict__ vh, const float* __restrict__ Asum,
    _Float16* __restrict__ oh) {
  extern __shared__ _Float16 smem[];
  _Float16* q2  = smem + EQ2;
  _Float16* k2  = smem + EK2;
  _Float16* k2T = smem + EK2T;
  _Float16* vT  = smem + EVT;
  _Float16* sc  = smem + ESC;
  _Float16* S0T = smem + ES0T;

  const int bh = blockIdx.x;
  const int h = bh & (H_ - 1);
  const int b = bh >> 4;
  const int tid = threadIdx.x;
  const int wave = tid >> 5, lane = tid & 31;

  for (int i = tid; i < 64 * LDH; i += 256) S0T[i] = (_Float16)0.f;
  __syncthreads();

  const int row = tid >> 2;  // 0..63
  const int seg = tid & 3;   // 0..3, 16 halfs each

  for (int c = 0; c < NC_; ++c) {
    // ---- stage chunk via Tensor Data Mover (wave 0 issues, all wait) ----
    const size_t gbase0 = (((size_t)b * S_ + (size_t)c * 64) * H_ + h) * DH_;
    if (wave == 0) {
      tdm_load_tile(qh + gbase0, (unsigned)(EQ2 * 2));
      tdm_load_tile(kh + gbase0, (unsigned)(EK2 * 2));
      tdm_load_tile(vh + gbase0, (unsigned)(ESC * 2));  // V rows staged in sc temp
      __builtin_amdgcn_s_wait_tensorcnt(0);
    }
    __syncthreads();

    // ---- build transposed copies from LDS (k2T, vT) ----
    {
      v8h lo = *(const v8h*)(k2 + row * LDH + seg * 16);
      v8h hi = *(const v8h*)(k2 + row * LDH + seg * 16 + 8);
#pragma unroll
      for (int i = 0; i < 8; ++i) {
        k2T[(seg * 16 + i) * LDH + row] = lo[i];
        k2T[(seg * 16 + 8 + i) * LDH + row] = hi[i];
      }
    }
    {
      v8h lo = *(const v8h*)(sc + row * LDH + seg * 16);
      v8h hi = *(const v8h*)(sc + row * LDH + seg * 16 + 8);
#pragma unroll
      for (int i = 0; i < 8; ++i) {
        vT[(seg * 16 + i) * LDH + row] = lo[i];
        vT[(seg * 16 + 8 + i) * LDH + row] = hi[i];
      }
    }
    __syncthreads();

    // ---- scores = q2 @ k2^T, causal mask, to LDS ----
#pragma unroll
    for (int it = 0; it < 2; ++it) {
      const int tt = wave * 2 + it, mi = tt >> 2, ni = tt & 3;
      v8f acc = {};
#pragma unroll
      for (int k0 = 0; k0 < 64; k0 += 32) {
        v16h a = load_a16x32(q2 + mi * 16 * LDH + k0, LDH, lane);
        v16h bm = load_b16x32(k2 + ni * 16 * LDH + k0, LDH, lane);
        acc = WMMA_F16(a, bm, acc);
      }
      const int n = lane & 15, hh = lane >> 4;
      const int jg = ni * 16 + n;
#pragma unroll
      for (int r = 0; r < 8; ++r) {
        const int tg = mi * 16 + hh * 8 + r;
        sc[tg * LDH + jg] = (jg <= tg) ? (_Float16)acc[r] : (_Float16)0.f;
      }
    }
    __syncthreads();

    // ---- O = q2 @ S0 + scores @ V ----
#pragma unroll
    for (int it = 0; it < 2; ++it) {
      const int tt = wave * 2 + it, mi = tt >> 2, ni = tt & 3;
      v8f acc = {};
#pragma unroll
      for (int k0 = 0; k0 < 64; k0 += 32) {  // inter-chunk: B[k=dk][n=dv] = S0T[n][k]
        v16h a = load_a16x32(q2 + mi * 16 * LDH + k0, LDH, lane);
        v16h bm = load_b16x32(S0T + ni * 16 * LDH + k0, LDH, lane);
        acc = WMMA_F16(a, bm, acc);
      }
#pragma unroll
      for (int k0 = 0; k0 < 64; k0 += 32) {  // intra-chunk: B[k=j][n=dv] = vT[n][k]
        v16h a = load_a16x32(sc + mi * 16 * LDH + k0, LDH, lane);
        v16h bm = load_b16x32(vT + ni * 16 * LDH + k0, LDH, lane);
        acc = WMMA_F16(a, bm, acc);
      }
      const int n = lane & 15, hh = lane >> 4;
#pragma unroll
      for (int r = 0; r < 8; ++r) {
        const int tg = c * 64 + mi * 16 + hh * 8 + r;
        const int dv = ni * 16 + n;
        oh[(((size_t)b * S_ + tg) * H_ + h) * DH_ + dv] = (_Float16)acc[r];
      }
    }
    __syncthreads();

    // ---- state: S^T <- exp(A_L) * (S^T + V^T @ K2) ----
    const float eAL = __expf(Asum[(b * H_ + h) * NC_ + c]);
#pragma unroll
    for (int it = 0; it < 2; ++it) {
      const int tt = wave * 2 + it, mi = tt >> 2, ni = tt & 3;
      v8f acc = {};
#pragma unroll
      for (int k0 = 0; k0 < 64; k0 += 32) {  // A[m=dv][k=t]=vT, B[k=t][n=dk]=k2T[n][k]
        v16h a = load_a16x32(vT + mi * 16 * LDH + k0, LDH, lane);
        v16h bm = load_b16x32(k2T + ni * 16 * LDH + k0, LDH, lane);
        acc = WMMA_F16(a, bm, acc);
      }
      const int n = lane & 15, hh = lane >> 4;
#pragma unroll
      for (int r = 0; r < 8; ++r) {
        const int dv = mi * 16 + hh * 8 + r, dk = ni * 16 + n;
        float old = (float)S0T[dv * LDH + dk];
        S0T[dv * LDH + dk] = (_Float16)(eAL * (old + acc[r]));
      }
    }
    __syncthreads();
  }
}

// ---------------- RMS norm * norm_w * sigmoid(og), in place on oh ----------
__global__ __launch_bounds__(256) void rms_og_gate_kernel(
    _Float16* __restrict__ o, const _Float16* __restrict__ ogh,
    const float* __restrict__ norm_w) {
  const int token = blockIdx.x;
  const int tid = threadIdx.x;
  __shared__ float red[8];
  __shared__ float rs_s;
  const size_t base = (size_t)token * HID_;
  float v[4];
  float ss = 0.f;
#pragma unroll
  for (int i = 0; i < 4; ++i) {
    v[i] = (float)o[base + tid + 256 * i];
    ss += v[i] * v[i];
  }
#pragma unroll
  for (int off = 16; off > 0; off >>= 1) ss += __shfl_down(ss, off, 32);
  if ((tid & 31) == 0) red[tid >> 5] = ss;
  __syncthreads();
  if (tid == 0) {
    float t = 0.f;
#pragma unroll
    for (int i = 0; i < 8; ++i) t += red[i];
    rs_s = rsqrtf(t / (float)HID_ + 1.1920929e-7f);
  }
  __syncthreads();
  const float rs = rs_s;
#pragma unroll
  for (int i = 0; i < 4; ++i) {
    const int col = tid + 256 * i;
    const float og = (float)ogh[base + col];
    const float sig = 1.f / (1.f + __expf(-og));
    o[base + col] = (_Float16)(v[i] * rs * norm_w[col] * sig);
  }
}

// ---------------- host ----------------
extern "C" void kernel_launch(void* const* d_in, const int* in_sizes, int n_in,
                              void* d_out, int out_size, void* d_ws, size_t ws_size,
                              hipStream_t stream) {
  const float* x   = (const float*)d_in[0];
  const float* Wq  = (const float*)d_in[1];
  const float* Wk  = (const float*)d_in[2];
  const float* Wv  = (const float*)d_in[3];
  const float* Wo  = (const float*)d_in[4];
  const float* Wg  = (const float*)d_in[5];
  const float* Wog = (const float*)d_in[6];
  const float* Wd  = (const float*)d_in[7];
  const float* bd  = (const float*)d_in[8];
  const float* nw  = (const float*)d_in[9];
  (void)in_sizes; (void)n_in; (void)out_size; (void)ws_size;

  char* p = (char*)d_ws;
  auto alloc = [&](size_t bytes) -> void* {
    void* r = (void*)p;
    p += (bytes + 255) & ~(size_t)255;
    return r;
  };

  _Float16* xh   = (_Float16*)alloc((size_t)MTOK_ * DM_ * 2);
  _Float16* wqh  = (_Float16*)alloc((size_t)HID_ * DM_ * 2);
  _Float16* wkh  = (_Float16*)alloc((size_t)HID_ * DM_ * 2);
  _Float16* wvh  = (_Float16*)alloc((size_t)HID_ * DM_ * 2);
  _Float16* wgh  = (_Float16*)alloc((size_t)HID_ * DM_ * 2);
  _Float16* wogh = (_Float16*)alloc((size_t)HID_ * DM_ * 2);
  _Float16* woh  = (_Float16*)alloc((size_t)DM_ * HID_ * 2);
  _Float16* wdh  = (_Float16*)alloc((size_t)H_ * DM_ * 2);
  _Float16* qh   = (_Float16*)alloc((size_t)MTOK_ * HID_ * 2);
  _Float16* kh   = (_Float16*)alloc((size_t)MTOK_ * HID_ * 2);
  _Float16* vh   = (_Float16*)alloc((size_t)MTOK_ * HID_ * 2);
  _Float16* gh   = (_Float16*)alloc((size_t)MTOK_ * HID_ * 2);
  _Float16* ogh  = (_Float16*)alloc((size_t)MTOK_ * HID_ * 2);
  float*    dpr  = (float*)alloc((size_t)MTOK_ * H_ * 4);
  float*    Asum = (float*)alloc((size_t)B_ * H_ * NC_ * 4);
  _Float16* oh   = (_Float16*)alloc((size_t)MTOK_ * HID_ * 2);

  auto cvt = [&](const float* in, _Float16* out, int n) {
    cvt_f32_to_f16_kernel<<<(n + 255) / 256, 256, 0, stream>>>(in, out, n);
  };
  cvt(x,   xh,   MTOK_ * DM_);
  cvt(Wq,  wqh,  HID_ * DM_);
  cvt(Wk,  wkh,  HID_ * DM_);
  cvt(Wv,  wvh,  HID_ * DM_);
  cvt(Wg,  wgh,  HID_ * DM_);
  cvt(Wog, wogh, HID_ * DM_);
  cvt(Wo,  woh,  DM_ * HID_);
  cvt(Wd,  wdh,  H_ * DM_);

  dim3 gBig(HID_ / 64, MTOK_ / 128);   // (16, 32)
  wmma_gemm_kernel<true><<<gBig, 256, 0, stream>>>(xh, wqh,  nullptr, qh,  MTOK_, HID_, DM_);
  wmma_gemm_kernel<true><<<gBig, 256, 0, stream>>>(xh, wkh,  nullptr, kh,  MTOK_, HID_, DM_);
  wmma_gemm_kernel<true><<<gBig, 256, 0, stream>>>(xh, wvh,  nullptr, vh,  MTOK_, HID_, DM_);
  wmma_gemm_kernel<true><<<gBig, 256, 0, stream>>>(xh, wgh,  nullptr, gh,  MTOK_, HID_, DM_);
  wmma_gemm_kernel<true><<<gBig, 256, 0, stream>>>(xh, wogh, nullptr, ogh, MTOK_, HID_, DM_);
  dim3 gD(1, MTOK_ / 128);
  wmma_gemm_kernel<false><<<gD, 256, 0, stream>>>(xh, wdh, dpr, nullptr, MTOK_, H_, DM_);

  gla_prep_kernel<<<B_ * H_ * NC_, 64, 0, stream>>>(qh, kh, gh, dpr, bd, Asum);
  gla_scan_kernel<<<B_ * H_, 256, SCAN_LDS_BYTES, stream>>>(qh, kh, vh, Asum, oh);
  rms_og_gate_kernel<<<MTOK_, 256, 0, stream>>>(oh, ogh, nw);

  dim3 gOut(DM_ / 64, MTOK_ / 128);
  wmma_gemm_kernel<false><<<gOut, 256, 0, stream>>>(oh, woh, (float*)d_out, nullptr,
                                                    MTOK_, DM_, HID_);
}
